// CausalSelfAttention_86861418594520
// MI455X (gfx1250) — compile-verified, matrix operand order checked by reading, not run on hardware
//
#include <hip/hip_runtime.h>
#include <hip/hip_bf16.h>

// ---------------------------------------------------------------------------
// Causal self-attention forward, all GEMMs via v_wmma_f32_16x16x32_bf16.
// x:(4,2048,1024) f32, w_qkv:(1024,3072), b_qkv:(3072), w_out:(1024,1024),
// b_out:(1024)  ->  out:(4,2048,1024) f32
// ---------------------------------------------------------------------------

typedef __attribute__((ext_vector_type(16))) __bf16 v16bf;
typedef __attribute__((ext_vector_type(8)))  float  v8f;

#define BATCH 4
#define SEQ   2048
#define DMODEL 1024
#define NHEAD 16
#define HDIM  64
#define N3    3072

// Lane's A/B-operand fragment for wmma_f32_16x16x32_bf16:
// lane<16 holds K = 0..7 and 16..23 ; lane>=16 holds K = 8..15 and 24..31.
// Caller passes p = row_base + k0 + hs  (hs = (lane&16)?8:0); chunks at p, p+16.
__device__ __forceinline__ v16bf load_tile_a_bf16(const __bf16* p) {
  union { uint4 u; __bf16 h[8]; } a, b;
  a.u = *(const uint4*)p;
  b.u = *(const uint4*)(p + 16);
  v16bf r;
#pragma unroll
  for (int i = 0; i < 8; i++) { r[i] = a.h[i]; r[i + 8] = b.h[i]; }
  return r;
}

__device__ __forceinline__ v8f wmma_bf16(v16bf a, v16bf b, v8f c) {
  return __builtin_amdgcn_wmma_f32_16x16x32_bf16(false, a, false, b, (short)0, c,
                                                 false, false);
}

// ---------------------------------------------------------------------------
// src: (K x N) row-major f32  ->  dst: (N x K) bf16 (N-major) for B operands.
__global__ void cvt_transpose_bf16(const float* __restrict__ src,
                                   __bf16* __restrict__ dst, int K, int N) {
  int idx = blockIdx.x * blockDim.x + threadIdx.x;
  if (idx < K * N) {
    int n = idx / K, k = idx - n * K;
    dst[idx] = (__bf16)src[(size_t)k * N + n];
  }
}

// ---------------------------------------------------------------------------
// Straight f32 -> bf16 convert, 8 elems/thread, fully coalesced.
__global__ void cvt_bf16(const float* __restrict__ src, __bf16* __restrict__ dst,
                         int n) {
  int i = (blockIdx.x * blockDim.x + threadIdx.x) * 8;
  if (i < n) {
    float4 a = *(const float4*)(src + i);
    float4 b = *(const float4*)(src + i + 4);
    __bf16 o[8] = {(__bf16)a.x, (__bf16)a.y, (__bf16)a.z, (__bf16)a.w,
                   (__bf16)b.x, (__bf16)b.y, (__bf16)b.z, (__bf16)b.w};
    *(uint4*)(dst + i) = *(uint4*)o;
  }
}

// ---------------------------------------------------------------------------
// QKV = xbf @ w_qkv + b, scattered into Q,K (b,h,s,hd) bf16 and Vt (b,h,hd,s).
// Block = 128 thr (4 waves). Wave tile 32x64; block tile 64x128.
__global__ __launch_bounds__(128) void qkv_gemm(
    const __bf16* __restrict__ xbf, const __bf16* __restrict__ wT,
    const float* __restrict__ bias, __bf16* __restrict__ Q,
    __bf16* __restrict__ Kd, __bf16* __restrict__ Vt) {
  const int lane = threadIdx.x & 31;
  const int wave = threadIdx.x >> 5;
  const int hs = (lane & 16) ? 8 : 0;
  const int lr = lane & 15;
  const int m0 = blockIdx.y * 64 + (wave >> 1) * 32;
  const int n0 = blockIdx.x * 128 + (wave & 1) * 64;

  v8f acc[2][4] = {};
  for (int k0 = 0; k0 < DMODEL; k0 += 32) {
    v16bf A[2], B[4];
#pragma unroll
    for (int i = 0; i < 2; i++)
      A[i] = load_tile_a_bf16(xbf + (size_t)(m0 + 16 * i + lr) * DMODEL + k0 + hs);
#pragma unroll
    for (int j = 0; j < 4; j++)
      B[j] = load_tile_a_bf16(wT + (size_t)(n0 + 16 * j + lr) * DMODEL + k0 + hs);
#pragma unroll
    for (int i = 0; i < 2; i++)
#pragma unroll
      for (int j = 0; j < 4; j++) acc[i][j] = wmma_bf16(A[i], B[j], acc[i][j]);
  }

#pragma unroll
  for (int i = 0; i < 2; i++)
#pragma unroll
    for (int j = 0; j < 4; j++)
#pragma unroll
      for (int r = 0; r < 8; r++) {
        int row = m0 + 16 * i + r + hs;          // C layout: M = r + 8*(lane>=16)
        int col = n0 + 16 * j + lr;              // N = lane%16
        float v = acc[i][j][r] + bias[col];
        int b = row >> 11, s = row & (SEQ - 1);
        int which = col >> 10, rem = col & (DMODEL - 1);
        int h = rem >> 6, d = rem & (HDIM - 1);
        int bh = b * NHEAD + h;
        __bf16 bv = (__bf16)v;
        if (which == 0)      Q[((size_t)bh * SEQ + s) * HDIM + d] = bv;
        else if (which == 1) Kd[((size_t)bh * SEQ + s) * HDIM + d] = bv;
        else                 Vt[((size_t)bh * HDIM + d) * SEQ + s] = bv;
      }
}

// ---------------------------------------------------------------------------
// Flash attention. Block = 256 thr (8 waves); block owns (b,h) x 128 q-rows,
// each wave 16 q-rows. Key blocks of 64; LDS round-trip reshapes P (C-layout)
// into A-layout for the P@V WMMA.
__global__ __launch_bounds__(256) void attn(
    const __bf16* __restrict__ Q, const __bf16* __restrict__ Kd,
    const __bf16* __restrict__ Vt, __bf16* __restrict__ O) {
  __shared__ __bf16 sP[8][16][72];  // padded stride: 144B rows, dodge banks
  const int lane = threadIdx.x & 31;
  const int wave = threadIdx.x >> 5;
  const int hs = (lane & 16) ? 8 : 0;
  const int lr = lane & 15;
  const int bh = blockIdx.y;
  const int b = bh >> 4, h = bh & 15;
  const int q0 = blockIdx.x * 128;
  const int qbase = q0 + wave * 16;
  const __bf16* Qb = Q + (size_t)bh * SEQ * HDIM;
  const __bf16* Kb = Kd + (size_t)bh * SEQ * HDIM;
  const __bf16* Vb = Vt + (size_t)bh * HDIM * SEQ;

  v16bf Qa[2];
#pragma unroll
  for (int t = 0; t < 2; t++)
    Qa[t] = load_tile_a_bf16(Qb + (size_t)(qbase + lr) * HDIM + 32 * t + hs);

  v8f Oacc[4] = {};
  float m_r[8], l_r[8];
#pragma unroll
  for (int r = 0; r < 8; r++) { m_r[r] = -1e30f; l_r[r] = 0.f; }

  const float LOG2E = 1.4426950408889634f;
  const int kend = q0 + 128;  // uniform trip count across all waves in block

  for (int kb = 0; kb < kend; kb += 64) {
    // S = Q @ K^T  (B column n = key -> contiguous K row)
    v8f Sacc[4] = {};
#pragma unroll
    for (int jk = 0; jk < 4; jk++)
#pragma unroll
      for (int t = 0; t < 2; t++) {
        v16bf Bt = load_tile_a_bf16(Kb + (size_t)(kb + 16 * jk + lr) * HDIM +
                                    32 * t + hs);
        Sacc[jk] = wmma_bf16(Qa[t], Bt, Sacc[jk]);
      }
    // scale + causal mask (branchless, EXEC stays full)
#pragma unroll
    for (int jk = 0; jk < 4; jk++) {
      int key = kb + 16 * jk + lr;
#pragma unroll
      for (int r = 0; r < 8; r++) {
        int q = qbase + r + hs;
        float s = Sacc[jk][r] * 0.125f;  // hd^-0.5
        Sacc[jk][r] = (key <= q) ? s : -1e30f;
      }
    }
    // online softmax: per-row max/sum reductions across each 16-lane half
    float alpha[8];
#pragma unroll
    for (int r = 0; r < 8; r++) {
      float v = fmaxf(fmaxf(Sacc[0][r], Sacc[1][r]),
                      fmaxf(Sacc[2][r], Sacc[3][r]));
#pragma unroll
      for (int msk = 1; msk < 16; msk <<= 1) v = fmaxf(v, __shfl_xor(v, msk, 32));
      float mn = fmaxf(m_r[r], v);
      alpha[r] = exp2f((m_r[r] - mn) * LOG2E);
      m_r[r] = mn;
    }
    float ps[8];
#pragma unroll
    for (int r = 0; r < 8; r++) ps[r] = 0.f;
#pragma unroll
    for (int jk = 0; jk < 4; jk++)
#pragma unroll
      for (int r = 0; r < 8; r++) {
        float p = exp2f((Sacc[jk][r] - m_r[r]) * LOG2E);
        ps[r] += p;
        sP[wave][r + hs][16 * jk + lr] = (__bf16)p;  // C-layout -> LDS
      }
#pragma unroll
    for (int r = 0; r < 8; r++) {
#pragma unroll
      for (int msk = 1; msk < 16; msk <<= 1) ps[r] += __shfl_xor(ps[r], msk, 32);
      l_r[r] = l_r[r] * alpha[r] + ps[r];
#pragma unroll
      for (int j = 0; j < 4; j++) Oacc[j][r] *= alpha[r];
    }
    asm volatile("s_wait_dscnt 0" ::: "memory");  // LDS RAW within wave
    v16bf Pa[2];
#pragma unroll
    for (int t = 0; t < 2; t++)
      Pa[t] = load_tile_a_bf16(&sP[wave][lr][0] + 32 * t + hs);
    // O += P @ V   (B column n = d -> contiguous Vt row)
#pragma unroll
    for (int j = 0; j < 4; j++)
#pragma unroll
      for (int t = 0; t < 2; t++) {
        v16bf Bv = load_tile_a_bf16(Vb + (size_t)(16 * j + lr) * SEQ + kb +
                                    32 * t + hs);
        Oacc[j] = wmma_bf16(Pa[t], Bv, Oacc[j]);
      }
    asm volatile("s_wait_dscnt 0" ::: "memory");  // WAR before next iter stores
  }

  float inv[8];
#pragma unroll
  for (int r = 0; r < 8; r++) inv[r] = 1.0f / l_r[r];
#pragma unroll
  for (int j = 0; j < 4; j++)
#pragma unroll
    for (int r = 0; r < 8; r++) {
      int q = qbase + r + hs;
      O[((size_t)b * SEQ + q) * DMODEL + h * HDIM + 16 * j + lr] =
          (__bf16)(Oacc[j][r] * inv[r]);
    }
}

// ---------------------------------------------------------------------------
// out = O @ w_out + b_out (f32 output). Same tiling as qkv_gemm.
__global__ __launch_bounds__(128) void out_gemm(
    const __bf16* __restrict__ A, const __bf16* __restrict__ wT,
    const float* __restrict__ bias, float* __restrict__ out) {
  const int lane = threadIdx.x & 31;
  const int wave = threadIdx.x >> 5;
  const int hs = (lane & 16) ? 8 : 0;
  const int lr = lane & 15;
  const int m0 = blockIdx.y * 64 + (wave >> 1) * 32;
  const int n0 = blockIdx.x * 128 + (wave & 1) * 64;

  v8f acc[2][4] = {};
  for (int k0 = 0; k0 < DMODEL; k0 += 32) {
    v16bf Av[2], Bv[4];
#pragma unroll
    for (int i = 0; i < 2; i++)
      Av[i] = load_tile_a_bf16(A + (size_t)(m0 + 16 * i + lr) * DMODEL + k0 + hs);
#pragma unroll
    for (int j = 0; j < 4; j++)
      Bv[j] = load_tile_a_bf16(wT + (size_t)(n0 + 16 * j + lr) * DMODEL + k0 + hs);
#pragma unroll
    for (int i = 0; i < 2; i++)
#pragma unroll
      for (int j = 0; j < 4; j++) acc[i][j] = wmma_bf16(Av[i], Bv[j], acc[i][j]);
  }
#pragma unroll
  for (int i = 0; i < 2; i++)
#pragma unroll
    for (int j = 0; j < 4; j++)
#pragma unroll
      for (int r = 0; r < 8; r++) {
        int row = m0 + 16 * i + r + hs;
        int col = n0 + 16 * j + lr;
        out[(size_t)row * DMODEL + col] = acc[i][j][r] + bias[col];
      }
}

// ---------------------------------------------------------------------------
extern "C" void kernel_launch(void* const* d_in, const int* in_sizes, int n_in,
                              void* d_out, int out_size, void* d_ws,
                              size_t ws_size, hipStream_t stream) {
  const float* x     = (const float*)d_in[0];
  const float* w_qkv = (const float*)d_in[1];
  const float* b_qkv = (const float*)d_in[2];
  const float* w_out = (const float*)d_in[3];
  const float* b_out = (const float*)d_in[4];
  float* out = (float*)d_out;

  char* ws = (char*)d_ws;
  size_t off = 0;
  __bf16* wqkvT = (__bf16*)(ws + off); off += (size_t)N3 * DMODEL * 2;      // 6 MB
  __bf16* woutT = (__bf16*)(ws + off); off += (size_t)DMODEL * DMODEL * 2;  // 2 MB
  __bf16* Q  = (__bf16*)(ws + off); off += (size_t)BATCH * NHEAD * SEQ * HDIM * 2;
  __bf16* K  = (__bf16*)(ws + off); off += (size_t)BATCH * NHEAD * SEQ * HDIM * 2;
  __bf16* Vt = (__bf16*)(ws + off); off += (size_t)BATCH * NHEAD * SEQ * HDIM * 2;
  __bf16* O  = (__bf16*)(ws + off); off += (size_t)BATCH * SEQ * DMODEL * 2;
  // xbf aliases O: xbf is consumed by qkv_gemm, O is produced by attn which
  // runs strictly later on the same stream -> disjoint lifetimes, same size.
  __bf16* xbf = O;
  (void)ws_size; (void)in_sizes; (void)n_in; (void)out_size;

  {
    int n = DMODEL * N3;
    cvt_transpose_bf16<<<(n + 255) / 256, 256, 0, stream>>>(w_qkv, wqkvT, DMODEL, N3);
  }
  {
    int n = DMODEL * DMODEL;
    cvt_transpose_bf16<<<(n + 255) / 256, 256, 0, stream>>>(w_out, woutT, DMODEL, DMODEL);
  }
  {
    int n = BATCH * SEQ * DMODEL;  // 8,388,608 elements, 8 per thread
    cvt_bf16<<<(n / 8 + 255) / 256, 256, 0, stream>>>(x, xbf, n);
  }
  qkv_gemm<<<dim3(N3 / 128, (BATCH * SEQ) / 64), 128, 0, stream>>>(
      xbf, wqkvT, b_qkv, Q, K, Vt);
  attn<<<dim3(SEQ / 128, BATCH * NHEAD), 256, 0, stream>>>(Q, K, Vt, O);
  out_gemm<<<dim3(DMODEL / 128, (BATCH * SEQ) / 64), 128, 0, stream>>>(
      O, woutT, b_out, out);
}
